// Attention_13134009991266
// MI455X (gfx1250) — matrix operand order repaired and checked by reading, not compiled, not run on hardware
//
#include <hip/hip_runtime.h>

// ---------------------------------------------------------------------------
// Problem constants
// ---------------------------------------------------------------------------
#define B_      16
#define L_      1024
#define DMODEL  512
#define NHEADS  8
#define DHEAD   64
#define NUMREL  3969
#define MTOT    (B_ * L_)          // 16384 rows

typedef __attribute__((ext_vector_type(16))) __bf16 bf16x16;
typedef __attribute__((ext_vector_type(8)))  float  f32x8;
typedef __attribute__((ext_vector_type(4)))  unsigned int u32x4;
typedef __attribute__((ext_vector_type(8)))  int    i32x8;
typedef __attribute__((ext_vector_type(4)))  int    i32x4;

#if __has_builtin(__builtin_amdgcn_tensor_load_to_lds) && \
    __has_builtin(__builtin_amdgcn_s_wait_tensorcnt)
#define USE_TDM 1
#endif

// ---------------------------------------------------------------------------
// WMMA helpers — wave32 operand loads in this file's contiguous k order (k = 16*half + s)
// ---------------------------------------------------------------------------
__device__ __forceinline__ f32x8 wmma_bf16(bf16x16 a, bf16x16 b, f32x8 c) {
    return __builtin_amdgcn_wmma_f32_16x16x32_bf16(
        false, a, false, b, (short)0, c, false, false);
}

// A-matrix 16x32 (MxK): lane 0-15 -> M=lane, hi-half lanes K+16;
// VGPR j holds K = j*2 (+16 hi half). Two b128 runs per lane.
__device__ __forceinline__ bf16x16 ldsA(const __bf16* base, int ld) {
    const int lane = threadIdx.x & 31;
    const int m    = lane & 15;
    const int kh   = (lane >> 4) << 4;     // 0 or 16
    const __bf16* p = base + m * ld + kh;
    bf16x16 a;
#pragma unroll
    for (int j = 0; j < 8; ++j) {
        const int k = j << 1;
        a[2 * j]     = p[k];
        a[2 * j + 1] = p[k + 1];
    }
    return a;
}

// B-matrix 32x16 (KxN), K CONTIGUOUS per column: element (k,n) at
// base[n*nstride + k]; lane -> N=lane&15, K halves 0..15/16..31.
// Each lane reads 16 contiguous bf16 -> 2x ds_load_b128.
__device__ __forceinline__ bf16x16 ldsBc(const __bf16* base, int nstride) {
    const int lane = threadIdx.x & 31;
    const int n    = lane & 15;
    const int kh   = (lane >> 4) << 4;     // 0 or 16
    const __bf16* p = base + n * nstride + kh;
    bf16x16 b;
#pragma unroll
    for (int j = 0; j < 16; ++j) b[j] = p[j];
    return b;
}

// ---------------------------------------------------------------------------
// Tensor Data Mover: 2D tile (2-byte elements) global -> LDS.
// D# layout per ISA 8.3/8.4 (group0 128b, group1 256b; groups 2/3 unused).
// 6-arg builtin form: (g0, g1, g2:i32x4, g3:i32x4, g4:i32x8, cpol)
// ---------------------------------------------------------------------------
#ifdef USE_TDM
__device__ __forceinline__ void tdm_load_2d(unsigned int lds_off, unsigned long long gaddr,
                                            unsigned int tile_d0, unsigned int tile_d1,
                                            unsigned int tsr_d0, unsigned int tsr_d1,
                                            unsigned long long d0_stride) {
    u32x4 g0;
    g0[0] = 1u;                                              // count=1 (valid), user mode
    g0[1] = lds_off;                                         // lds_addr (bytes)
    g0[2] = (unsigned int)(gaddr & 0xffffffffu);             // global_addr[31:0]
    g0[3] = (unsigned int)((gaddr >> 32) & 0x01ffffffu)      // global_addr[56:32]
          | (2u << 30);                                      // type=2 ("image")
    i32x8 g1;
    g1[0] = (int)(1u << 16);                                 // data_size=1 (2-byte elems)
    g1[1] = (int)((tsr_d0 & 0xffffu) << 16);                 // tensor_dim0[15:0]
    g1[2] = (int)((tsr_d0 >> 16) | ((tsr_d1 & 0xffffu) << 16)); // dim0 hi | dim1 lo
    g1[3] = (int)((tsr_d1 >> 16) | (tile_d0 << 16));         // dim1 hi | tile_dim0
    g1[4] = (int)(tile_d1 & 0xffffu);                        // tile_dim1 (0 = unused)
    g1[5] = (int)(unsigned int)(d0_stride & 0xffffffffu);    // tensor_dim0_stride lo
    g1[6] = (int)(unsigned int)((d0_stride >> 32) & 0xffffu);// stride hi16
    g1[7] = 0;
    i32x4 z4 = {0, 0, 0, 0};
    i32x8 z8 = {0, 0, 0, 0, 0, 0, 0, 0};
    __builtin_amdgcn_tensor_load_to_lds(g0, g1, z4, z4, z8, 0);
}
#endif

// ---------------------------------------------------------------------------
// float -> bf16 cast
// ---------------------------------------------------------------------------
__global__ void k_cvt(const float* __restrict__ src, __bf16* __restrict__ dst, int n) {
    int i = blockIdx.x * blockDim.x + threadIdx.x;
    if (i < n) dst[i] = (__bf16)src[i];
}

// ---------------------------------------------------------------------------
// Fused QKV GEMM: Y = X @ W^T, double-buffered LDS (one barrier per K-step).
// W row-major [n][k] (K-contiguous B tiles). Q,K written [b][h][l][d];
// V written TRANSPOSED [b][h][d][l] (PV B-operand key-contiguous, and the
// epilogue becomes one 16B store per tile).
// ---------------------------------------------------------------------------
__global__ __launch_bounds__(256)
void k_gemm_qkv(const __bf16* __restrict__ X,
                const __bf16* __restrict__ Wq, const __bf16* __restrict__ Wk,
                const __bf16* __restrict__ Wv,
                __bf16* __restrict__ Q, __bf16* __restrict__ K,
                __bf16* __restrict__ V) {
    __shared__ __bf16 As[2][128 * 32];   // [m][k]
    __shared__ __bf16 Bs[2][128 * 32];   // [n][k]

    const int mbase = blockIdx.x * 128;
    const int nbase = blockIdx.y * 128;
    const int mat   = blockIdx.z;
    const __bf16* W = (mat == 0) ? Wq : (mat == 1) ? Wk : Wv;
    __bf16* Out     = (mat == 0) ? Q  : (mat == 1) ? K  : V;
    const float scale = (mat == 0) ? 0.125f : 1.0f;   // Q pre-scaled by 1/sqrt(64)

    const int tid  = threadIdx.x;
    const int wave = tid >> 5;
    const int wm   = (wave >> 2) * 64;
    const int wn   = (wave & 3) * 32;
    const int lr   = tid >> 3;          // 0..31
    const int lc   = (tid & 7) * 4;     // 0..28

    auto load_tile = [&](int bufi, int k0) {
#pragma unroll
        for (int p = 0; p < 4; ++p) {
            int r = lr + p * 32;
            *(uint2*)&As[bufi][r * 32 + lc] =
                *(const uint2*)&X[(size_t)(mbase + r) * DMODEL + k0 + lc];
            *(uint2*)&Bs[bufi][r * 32 + lc] =
                *(const uint2*)&W[(size_t)(nbase + r) * DMODEL + k0 + lc];
        }
    };

    const f32x8 vzero = {0.f, 0.f, 0.f, 0.f, 0.f, 0.f, 0.f, 0.f};
    f32x8 acc[4][2];
#pragma unroll
    for (int i = 0; i < 4; ++i)
#pragma unroll
        for (int j = 0; j < 2; ++j) acc[i][j] = vzero;

    load_tile(0, 0);
    int buf = 0;
    for (int k0 = 0; k0 < DMODEL; k0 += 32) {
        __syncthreads();                       // tile [buf] ready; [buf^1] free
        if (k0 + 32 < DMODEL) load_tile(buf ^ 1, k0 + 32);

        bf16x16 af[4];
#pragma unroll
        for (int i = 0; i < 4; ++i) af[i] = ldsA(&As[buf][(wm + i * 16) * 32], 32);
#pragma unroll
        for (int j = 0; j < 2; ++j) {
            bf16x16 bfr = ldsBc(&Bs[buf][(wn + j * 16) * 32], 32);
#pragma unroll
            for (int i = 0; i < 4; ++i) acc[i][j] = wmma_bf16(af[i], bfr, acc[i][j]);
        }
        buf ^= 1;
    }

    const int lane = tid & 31;
    const int n16  = lane & 15;
    const int mh   = (lane >> 4) * 8;

    if (mat == 2) {            // V: [b][h][d][l] -> rows are l-contiguous, 16B stores
#pragma unroll
        for (int i = 0; i < 4; ++i) {
            int gm0 = mbase + wm + i * 16 + mh;       // 128-tile never crosses a batch
            int bb = gm0 >> 10, l0 = gm0 & 1023;
#pragma unroll
            for (int j = 0; j < 2; ++j) {
                int gn = nbase + wn + j * 16 + n16;
                int hh = gn >> 6, dd = gn & 63;
                union { __bf16 e[8]; uint4 v; } pk;
#pragma unroll
                for (int r = 0; r < 8; ++r) pk.e[r] = (__bf16)acc[i][j][r];
                *(uint4*)&Out[(((size_t)(bb * NHEADS + hh)) * DHEAD + dd) * L_ + l0] = pk.v;
            }
        }
    } else {                   // Q/K: [b][h][l][d], stride-64 b16 stores
#pragma unroll
        for (int i = 0; i < 4; ++i) {
            int gm0 = mbase + wm + i * 16 + mh;
            int bb = gm0 >> 10, l0 = gm0 & 1023;
#pragma unroll
            for (int j = 0; j < 2; ++j) {
                int gn = nbase + wn + j * 16 + n16;
                int hh = gn >> 6, dd = gn & 63;
                __bf16* p = &Out[(((size_t)(bb * NHEADS + hh)) * L_ + l0) * DHEAD + dd];
#pragma unroll
                for (int r = 0; r < 8; ++r)
                    p[(size_t)r * DHEAD] = (__bf16)(acc[i][j][r] * scale);
            }
        }
    }
}

// ---------------------------------------------------------------------------
// Flash attention with relative-position bias.
// Grid: (L/128, B*H). 8 waves; wave owns 16 query rows.
// K/V tiles double-buffered: TDM streams tile j+1 while waves run WMMA on
// tile j (TENSORcnt in-order => s_wait_tensorcnt 2 releases tile j).
// ---------------------------------------------------------------------------
__global__ __launch_bounds__(256)
void k_attn(const __bf16* __restrict__ Q, const __bf16* __restrict__ K,
            const __bf16* __restrict__ Vt, const float* __restrict__ bias,
            const int* __restrict__ coords, __bf16* __restrict__ O) {
    __shared__ __bf16 Qs[128 * 64];        // [q][d]
    __shared__ __bf16 Ks[2][128 * 64];     // [key][d] (d contiguous -> K^T frags)
    __shared__ __bf16 Vs[2][64 * 128];     // [d][key] (key contiguous -> PV frags)
    __shared__ float  biasS[NUMREL];
    __shared__ __bf16 Ps[8 * 16 * 128];    // per-wave P staging

    const int bh    = blockIdx.y;
    const int h     = bh & 7;
    const int b     = bh >> 3;
    const int qbase = blockIdx.x * 128;
    const int tid   = threadIdx.x;
    const int wave  = tid >> 5;
    const int lane  = tid & 31;
    const int n16   = lane & 15;
    const int half  = lane >> 4;

    const __bf16* Qg = Q  + (size_t)bh * L_ * DHEAD;
    const __bf16* Kg = K  + (size_t)bh * L_ * DHEAD;
    const __bf16* Vg = Vt + (size_t)bh * DHEAD * L_;   // [d][l]

    for (int i = tid; i < NUMREL; i += 256) biasS[i] = bias[h * NUMREL + i];

    {   // Q tile 128x64
        int r = tid >> 3, c = (tid & 7) * 8;
#pragma unroll
        for (int p = 0; p < 4; ++p)
            *(uint4*)&Qs[(r + p * 32) * 64 + c] =
                *(const uint4*)&Qg[(size_t)(qbase + r + p * 32) * DHEAD + c];
    }

#ifdef USE_TDM
    auto tdm_issue = [&](int tbuf, int tkb) {
        // K tile: contiguous 128x64 bf16 block -> 1D copy of 8192 elements
        tdm_load_2d((unsigned int)(uintptr_t)Ks[tbuf],
                    (unsigned long long)(uintptr_t)(Kg + (size_t)tkb * DHEAD),
                    8192, 0, 8192, 1, 8192);
        // V tile: 64 rows (d) x 128 cols (key), global row stride L_
        tdm_load_2d((unsigned int)(uintptr_t)Vs[tbuf],
                    (unsigned long long)(uintptr_t)(Vg + tkb),
                    128, 64, 128, 64, L_);
    };
    if (wave == 0) tdm_issue(0, 0);        // prime tile 0
#endif

    const f32x8 vzero = {0.f, 0.f, 0.f, 0.f, 0.f, 0.f, 0.f, 0.f};
    f32x8 oacc[4];
#pragma unroll
    for (int t = 0; t < 4; ++t) oacc[t] = vzero;
    float mrow[8], rsum[8];
#pragma unroll
    for (int r = 0; r < 8; ++r) { mrow[r] = -3.0e38f; rsum[r] = 0.0f; }

    __bf16* Pw = &Ps[wave * 16 * 128];
    int buf = 0;

    for (int j = 0; j < 8; ++j) {
        const int kb = j * 128;
        __syncthreads();                   // all reads of buf^1 finished

#ifdef USE_TDM
        if (wave == 0) {
            if (j < 7) {
                tdm_issue(buf ^ 1, kb + 128);              // stream next tile
                __builtin_amdgcn_s_wait_tensorcnt(2);      // tile j complete
            } else {
                __builtin_amdgcn_s_wait_tensorcnt(0);
            }
        }
        asm volatile("" ::: "memory");
        __syncthreads();                   // tile j visible to all waves
#else
        {
            int r = tid >> 3, c = (tid & 7) * 8;
#pragma unroll
            for (int p = 0; p < 4; ++p)
                *(uint4*)&Ks[buf][(r + p * 32) * 64 + c] =
                    *(const uint4*)&Kg[(size_t)(kb + r + p * 32) * DHEAD + c];
            int rv = tid >> 4, cv = (tid & 15) * 8;
#pragma unroll
            for (int p = 0; p < 4; ++p)
                *(uint4*)&Vs[buf][(rv + p * 16) * 128 + cv] =
                    *(const uint4*)&Vg[(size_t)(rv + p * 16) * L_ + kb + cv];
        }
        __syncthreads();
#endif

        // prefetch next coords slab (L2-resident gather stream)
        if (j < 7)
            __builtin_prefetch(&coords[(size_t)(qbase + wave * 16 + half * 8) * L_ +
                                       kb + 128 + n16], 0, 0);

        // S = Qw (16x64) @ K^T (64x128): 8 n-tiles x 2 k-steps
        f32x8 s[8];
#pragma unroll
        for (int t = 0; t < 8; ++t) s[t] = vzero;
#pragma unroll
        for (int kk = 0; kk < 64; kk += 32) {
            bf16x16 a = ldsA(&Qs[(wave * 16) * 64 + kk], 64);
#pragma unroll
            for (int t = 0; t < 8; ++t) {
                bf16x16 bfr = ldsBc(&Ks[buf][(t * 16) * 64 + kk], 64); // B[k][n]=K[n][k]
                s[t] = wmma_bf16(a, bfr, s[t]);
            }
        }

        // relative-position bias gather
#pragma unroll
        for (int t = 0; t < 8; ++t) {
#pragma unroll
            for (int r = 0; r < 8; ++r) {
                int grow = qbase + wave * 16 + half * 8 + r;
                int gcol = kb + t * 16 + n16;
                s[t][r] += biasS[coords[(size_t)grow * L_ + gcol]];
            }
        }

        // online softmax (row r of the C layout lives in VGPR element r)
#pragma unroll
        for (int r = 0; r < 8; ++r) {
            float tm = s[0][r];
#pragma unroll
            for (int t = 1; t < 8; ++t) tm = fmaxf(tm, s[t][r]);
#pragma unroll
            for (int m = 1; m < 16; m <<= 1) tm = fmaxf(tm, __shfl_xor(tm, m));
            float mnew = fmaxf(mrow[r], tm);
            float sc   = __expf(mrow[r] - mnew);
            mrow[r]    = mnew;
            float ls = 0.0f;
#pragma unroll
            for (int t = 0; t < 8; ++t) {
                float p = __expf(s[t][r] - mnew);
                ls += p;
                Pw[(half * 8 + r) * 128 + t * 16 + n16] = (__bf16)p;
            }
#pragma unroll
            for (int m = 1; m < 16; m <<= 1) ls += __shfl_xor(ls, m);
            rsum[r] = rsum[r] * sc + ls;
#pragma unroll
            for (int ot = 0; ot < 4; ++ot) oacc[ot][r] *= sc;
        }

        // O += P (16x128) @ V (128x64): 4 n-tiles x 4 k-steps
#pragma unroll
        for (int kk = 0; kk < 128; kk += 32) {
            bf16x16 a = ldsA(&Pw[kk], 128);
#pragma unroll
            for (int ot = 0; ot < 4; ++ot) {
                bf16x16 bfr = ldsBc(&Vs[buf][(ot * 16) * 128 + kk], 128);
                oacc[ot] = wmma_bf16(a, bfr, oacc[ot]);
            }
        }
        buf ^= 1;
    }

    // normalize + write [b][l][h*64+d] bf16 for the output GEMM
#pragma unroll
    for (int ot = 0; ot < 4; ++ot)
#pragma unroll
        for (int r = 0; r < 8; ++r) {
            int row = qbase + wave * 16 + half * 8 + r;
            float o = oacc[ot][r] / rsum[r];
            O[((size_t)(b * L_ + row)) * DMODEL + h * DHEAD + ot * 16 + n16] = (__bf16)o;
        }
}

// ---------------------------------------------------------------------------
// Output projection: out = Oattn @ Wo^T + b_o (f32), double-buffered LDS
// ---------------------------------------------------------------------------
__global__ __launch_bounds__(256)
void k_gemm_out(const __bf16* __restrict__ X, const __bf16* __restrict__ W,
                const float* __restrict__ bo, float* __restrict__ out) {
    __shared__ __bf16 As[2][128 * 32];
    __shared__ __bf16 Bs[2][128 * 32];

    const int mbase = blockIdx.x * 128;
    const int nbase = blockIdx.y * 128;
    const int tid   = threadIdx.x;
    const int wave  = tid >> 5;
    const int wm    = (wave >> 2) * 64;
    const int wn    = (wave & 3) * 32;
    const int lr    = tid >> 3;
    const int lc    = (tid & 7) * 4;

    auto load_tile = [&](int bufi, int k0) {
#pragma unroll
        for (int p = 0; p < 4; ++p) {
            int r = lr + p * 32;
            *(uint2*)&As[bufi][r * 32 + lc] =
                *(const uint2*)&X[(size_t)(mbase + r) * DMODEL + k0 + lc];
            *(uint2*)&Bs[bufi][r * 32 + lc] =
                *(const uint2*)&W[(size_t)(nbase + r) * DMODEL + k0 + lc];
        }
    };

    const f32x8 vzero = {0.f, 0.f, 0.f, 0.f, 0.f, 0.f, 0.f, 0.f};
    f32x8 acc[4][2];
#pragma unroll
    for (int i = 0; i < 4; ++i)
#pragma unroll
        for (int j = 0; j < 2; ++j) acc[i][j] = vzero;

    load_tile(0, 0);
    int buf = 0;
    for (int k0 = 0; k0 < DMODEL; k0 += 32) {
        __syncthreads();
        if (k0 + 32 < DMODEL) load_tile(buf ^ 1, k0 + 32);

        bf16x16 af[4];
#pragma unroll
        for (int i = 0; i < 4; ++i) af[i] = ldsA(&As[buf][(wm + i * 16) * 32], 32);
#pragma unroll
        for (int j = 0; j < 2; ++j) {
            bf16x16 bfr = ldsBc(&Bs[buf][(wn + j * 16) * 32], 32);
#pragma unroll
            for (int i = 0; i < 4; ++i) acc[i][j] = wmma_bf16(af[i], bfr, acc[i][j]);
        }
        buf ^= 1;
    }

    const int lane = tid & 31;
    const int n16  = lane & 15;
    const int mh   = (lane >> 4) * 8;
#pragma unroll
    for (int i = 0; i < 4; ++i)
#pragma unroll
        for (int j = 0; j < 2; ++j) {
            int gm0 = mbase + wm + i * 16 + mh;
            int gn  = nbase + wn + j * 16 + n16;
            float* p = &out[(size_t)gm0 * DMODEL + gn];
            float bias_n = bo[gn];
#pragma unroll
            for (int r = 0; r < 8; ++r)
                p[(size_t)r * DMODEL] = acc[i][j][r] + bias_n;
        }
}

// ---------------------------------------------------------------------------
// Launch
// ---------------------------------------------------------------------------
extern "C" void kernel_launch(void* const* d_in, const int* in_sizes, int n_in,
                              void* d_out, int out_size, void* d_ws, size_t ws_size,
                              hipStream_t stream) {
    const float* x        = (const float*)d_in[0];
    const float* rel_bias = (const float*)d_in[1];
    const int*   coords   = (const int*)d_in[2];
    const float* Wq       = (const float*)d_in[3];
    const float* Wk       = (const float*)d_in[4];
    const float* Wv       = (const float*)d_in[5];
    const float* Wo       = (const float*)d_in[6];
    const float* bo       = (const float*)d_in[7];
    float*       out      = (float*)d_out;

    const size_t SZ_X   = (size_t)MTOT * DMODEL * sizeof(__bf16);
    const size_t SZ_W   = (size_t)DMODEL * DMODEL * sizeof(__bf16);
    const size_t SZ_QKV = (size_t)B_ * NHEADS * L_ * DHEAD * sizeof(__bf16);

    char* ws = (char*)d_ws;
    __bf16* xb  = (__bf16*)ws;  ws += SZ_X;
    __bf16* wqb = (__bf16*)ws;  ws += SZ_W;
    __bf16* wkb = (__bf16*)ws;  ws += SZ_W;
    __bf16* wvb = (__bf16*)ws;  ws += SZ_W;
    __bf16* wob = (__bf16*)ws;  ws += SZ_W;
    __bf16* Qb  = (__bf16*)ws;  ws += SZ_QKV;
    __bf16* Kb  = (__bf16*)ws;  ws += SZ_QKV;
    __bf16* Vb  = (__bf16*)ws;  ws += SZ_QKV;   // transposed [b][h][d][l]
    __bf16* Ob  = (__bf16*)ws;  ws += SZ_X;

    // 1) bf16 casts (weights stay row-major)
    k_cvt<<<(MTOT * DMODEL) / 256, 256, 0, stream>>>(x, xb, MTOT * DMODEL);
    k_cvt<<<(DMODEL * DMODEL) / 256, 256, 0, stream>>>(Wq, wqb, DMODEL * DMODEL);
    k_cvt<<<(DMODEL * DMODEL) / 256, 256, 0, stream>>>(Wk, wkb, DMODEL * DMODEL);
    k_cvt<<<(DMODEL * DMODEL) / 256, 256, 0, stream>>>(Wv, wvb, DMODEL * DMODEL);
    k_cvt<<<(DMODEL * DMODEL) / 256, 256, 0, stream>>>(Wo, wob, DMODEL * DMODEL);

    // 2) fused QKV projection (Q pre-scaled, V transposed)
    k_gemm_qkv<<<dim3(MTOT / 128, DMODEL / 128, 3), 256, 0, stream>>>(
        xb, wqb, wkb, wvb, Qb, Kb, Vb);

    // 3) flash attention with rel-pos bias (TDM double-buffered K/V tiles)
    k_attn<<<dim3(L_ / 128, B_ * NHEADS), 256, 0, stream>>>(
        Qb, Kb, Vb, rel_bias, coords, Ob);

    // 4) output projection + bias
    k_gemm_out<<<dim3(MTOT / 128, DMODEL / 128), 256, 0, stream>>>(Ob, wob, bo, out);
}
